// Experts_4887672783477
// MI455X (gfx1250) — compile-verified, matrix operand order checked by reading, not run on hardware
//
#include <hip/hip_runtime.h>
#include <hip/hip_bf16.h>

// ---------------------------------------------------------------------------
// Per-expert SwiGLU FFN for MI455X (gfx1250), bf16 WMMA path.
//   og = (x@w_c_fc + b_fc) * silu(x@w_gate + b_gate)   [E,T,H]  (bf16 in ws)
//   o  = og @ w_c_proj + b_proj                          [E,T,D]  (f32 out)
// E=8, T=2048, D=1024, H=4096
// Double-buffered LDS, packed bf16 staging stores, async global->LDS copy
// for the bf16 A tile of the projection GEMM.
// ---------------------------------------------------------------------------

typedef __attribute__((ext_vector_type(16))) __bf16 v16bf;
typedef __attribute__((ext_vector_type(8)))  __bf16 v8bf;
typedef __attribute__((ext_vector_type(4)))  __bf16 v4bf;
typedef __attribute__((ext_vector_type(8)))  float  v8f;
typedef __attribute__((ext_vector_type(4)))  float  v4f;

constexpr int kE = 8;
constexpr int kT = 2048;
constexpr int kD = 1024;
constexpr int kH = 4096;

constexpr int BK  = 32;   // WMMA K per step (bf16 k=32)
constexpr int LDK = 40;   // LDS row stride in halves (32 + 8 pad -> 80B rows)

// Pack two f32 -> two bf16 in one dword (k-pair for transposed B tiles).
// Union form: clang fuses adjacent-source cases into v_cvt_pk_bf16_f32.
static __device__ inline unsigned pack2_bf16(float a, float b) {
    union { __bf16 h[2]; unsigned u; } p;
    p.h[0] = (__bf16)a; p.h[1] = (__bf16)b;
    return p.u;
}

// Async global -> LDS 128-bit copy (gfx1250 GLOBAL_LOAD_ASYNC_TO_LDS_B128).
// Low 32 bits of the generic LDS pointer are the LDS byte address.
static __device__ inline void async_copy_b128(__bf16* lds_dst,
                                              const __bf16* gsrc) {
    unsigned lds_addr = (unsigned)(size_t)lds_dst;
    asm volatile("global_load_async_to_lds_b128 %0, %1, off"
                 :: "v"(lds_addr), "v"(gsrc) : "memory");
}
static __device__ inline void wait_async() {
    asm volatile("s_wait_asynccnt 0x0" ::: "memory");
}

// Load a 16x32 bf16 fragment (A layout) from an LDS row base.
// Lane L holds row M = L%16; elements 0..7 are K=(L/16)*8+0..7,
// elements 8..15 are K=16+(L/16)*8+0..7.
static __device__ inline v16bf load_frag(const __bf16* row_base, int lhalf) {
    v8bf lo = *(const v8bf*)(row_base + lhalf * 8);
    v8bf hi = *(const v8bf*)(row_base + 16 + lhalf * 8);
    v16bf r;
#pragma unroll
    for (int i = 0; i < 8; ++i) { r[i] = lo[i]; r[i + 8] = hi[i]; }
    return r;
}

static __device__ inline v8f wmma_bf16(v16bf a, v16bf b, v8f c) {
    return __builtin_amdgcn_wmma_f32_16x16x32_bf16(
        false, a, false, b, (short)0, c, false, false);
}

// ---------------------------------------------------------------------------
// Kernel 1: dual GEMM (fc + gate) sharing the A tile, fused SwiGLU epilogue.
// Block tile: 128 (T) x 64 (H). 8 waves in 4x2; each wave owns 32x32.
// ---------------------------------------------------------------------------
__global__ __launch_bounds__(256)
void swiglu_dualgemm_k1(const float* __restrict__ x,
                        const float* __restrict__ w_fc,
                        const float* __restrict__ b_fc,
                        const float* __restrict__ w_gate,
                        const float* __restrict__ b_gate,
                        __bf16* __restrict__ og) {
    constexpr int BM = 128, BN = 64;
    __shared__ alignas(16) __bf16 As[2][BM * LDK];
    __shared__ alignas(16) __bf16 Bf[2][BN * LDK];
    __shared__ alignas(16) __bf16 Bg[2][BN * LDK];

    const int e  = blockIdx.z;
    const int m0 = blockIdx.x * BM;
    const int n0 = blockIdx.y * BN;

    const float* xe  = x      + (size_t)e * kT * kD;
    const float* wfe = w_fc   + (size_t)e * kD * kH;
    const float* wge = w_gate + (size_t)e * kD * kH;
    __bf16*      oge = og     + (size_t)e * kT * kH;

    const int tid   = threadIdx.x;
    const int lane  = tid & 31;
    const int wave  = tid >> 5;
    const int wm    = wave & 3;   // 0..3 -> 32-row slab
    const int wn    = wave >> 2;  // 0..1 -> 32-col slab
    const int l16   = lane & 15;
    const int lhalf = lane >> 4;

    // Staging maps (tile sizes divide exactly; offsets are loop-invariant).
    const int arow = tid >> 3;        // + it*32 -> 0..127
    const int ac4  = (tid & 7) * 4;   // k offset 0..28
    const int bn4  = (tid & 15) * 4;  // n offset 0..60
    const int br   = (tid >> 4) * 2;  // k-pair base 0..30

    v4f a_st[4], bf_st[2], bg_st[2];

    auto fetch = [&](int kb) {
#pragma unroll
        for (int it = 0; it < 4; ++it)
            a_st[it] = *(const v4f*)(xe + (size_t)(m0 + arow + it * 32) * kD + kb + ac4);
#pragma unroll
        for (int q = 0; q < 2; ++q) {
            bf_st[q] = *(const v4f*)(wfe + (size_t)(kb + br + q) * kH + n0 + bn4);
            bg_st[q] = *(const v4f*)(wge + (size_t)(kb + br + q) * kH + n0 + bn4);
        }
    };
    auto stage = [&](int s) {
#pragma unroll
        for (int it = 0; it < 4; ++it) {
            v4bf r;                    // adjacent elements -> v_cvt_pk fusion
#pragma unroll
            for (int j = 0; j < 4; ++j) r[j] = (__bf16)a_st[it][j];
            *(v4bf*)&As[s][(arow + it * 32) * LDK + ac4] = r;  // 8B ds_store
        }
#pragma unroll
        for (int j = 0; j < 4; ++j) {  // k-pair packed 4B ds_store
            *(unsigned*)&Bf[s][(bn4 + j) * LDK + br] = pack2_bf16(bf_st[0][j], bf_st[1][j]);
            *(unsigned*)&Bg[s][(bn4 + j) * LDK + br] = pack2_bf16(bg_st[0][j], bg_st[1][j]);
        }
    };

    v8f acc_f[2][2], acc_g[2][2];
#pragma unroll
    for (int i = 0; i < 2; ++i)
#pragma unroll
        for (int j = 0; j < 2; ++j) {
            acc_f[i][j] = (v8f)0.0f;
            acc_g[i][j] = (v8f)0.0f;
        }

    fetch(0);
    stage(0);
    int s = 0;
    for (int kb = 0; kb < kD; kb += BK, s ^= 1) {
        __syncthreads();
        const bool more = (kb + BK) < kD;
        if (more) fetch(kb + BK);      // global loads overlap WMMAs below

        v16bf afrag[2], bffrag[2], bgfrag[2];
#pragma unroll
        for (int i = 0; i < 2; ++i)
            afrag[i] = load_frag(&As[s][(wm * 32 + i * 16 + l16) * LDK], lhalf);
#pragma unroll
        for (int j = 0; j < 2; ++j) {
            bffrag[j] = load_frag(&Bf[s][(wn * 32 + j * 16 + l16) * LDK], lhalf);
            bgfrag[j] = load_frag(&Bg[s][(wn * 32 + j * 16 + l16) * LDK], lhalf);
        }
#pragma unroll
        for (int i = 0; i < 2; ++i)
#pragma unroll
            for (int j = 0; j < 2; ++j) {
                acc_f[i][j] = wmma_bf16(afrag[i], bffrag[j], acc_f[i][j]);
                acc_g[i][j] = wmma_bf16(afrag[i], bgfrag[j], acc_g[i][j]);
            }
        if (more) stage(s ^ 1);
    }

    // ---- Epilogue: biases + SwiGLU, store bf16 og ----
    float bfc[2], bgc[2];
#pragma unroll
    for (int j = 0; j < 2; ++j) {
        int col = n0 + wn * 32 + j * 16 + l16;
        bfc[j] = b_fc[(size_t)e * kH + col];
        bgc[j] = b_gate[(size_t)e * kH + col];
    }
#pragma unroll
    for (int i = 0; i < 2; ++i)
#pragma unroll
        for (int j = 0; j < 2; ++j) {
            int col = n0 + wn * 32 + j * 16 + l16;
#pragma unroll
            for (int r = 0; r < 8; ++r) {
                int row = m0 + wm * 32 + i * 16 + lhalf * 8 + r;
                float h = acc_f[i][j][r] + bfc[j];
                float g = acc_g[i][j][r] + bgc[j];
                float sil = g * (1.0f / (1.0f + __expf(-g)));
                oge[(size_t)row * kH + col] = (__bf16)(h * sil);
            }
        }
}

// ---------------------------------------------------------------------------
// Kernel 2: o = og(bf16) @ w_c_proj + b_proj.
// Block tile: 128 (T) x 128 (D). 8 waves in 2x4; each wave owns 64x32.
// A tile copied with GLOBAL_LOAD_ASYNC_TO_LDS_B128 (bf16, no conversion).
// ---------------------------------------------------------------------------
__global__ __launch_bounds__(256)
void proj_gemm_k2(const __bf16* __restrict__ og,
                  const float* __restrict__ w_proj,
                  const float* __restrict__ b_proj,
                  float* __restrict__ out) {
    constexpr int BM = 128, BN = 128;
    __shared__ alignas(16) __bf16 As[2][BM * LDK];
    __shared__ alignas(16) __bf16 Bs[2][BN * LDK];

    const int e  = blockIdx.z;
    const int m0 = blockIdx.x * BM;
    const int n0 = blockIdx.y * BN;

    const __bf16* oge = og     + (size_t)e * kT * kH;
    const float*  wpe = w_proj + (size_t)e * kH * kD;
    float*        oe  = out    + (size_t)e * kT * kD;

    const int tid   = threadIdx.x;
    const int lane  = tid & 31;
    const int wave  = tid >> 5;
    const int wm    = wave & 1;   // 0..1 -> 64-row slab
    const int wn    = wave >> 1;  // 0..3 -> 32-col slab
    const int l16   = lane & 15;
    const int lhalf = lane >> 4;

    // Staging maps.
    const int arow = tid >> 2;        // + it*64 -> 0..127
    const int ac8  = (tid & 3) * 8;   // k offset 0,8,16,24 (halves)
    const int bn4  = (tid & 31) * 4;  // n offset 0..124
    const int br   = (tid >> 5) * 4;  // k-quad base 0..28

    v4f b_st[4];

    // Issues async A-tile copy directly into LDS buffer s and fetches B rows.
    auto fetch = [&](int kb, int s) {
#pragma unroll
        for (int it = 0; it < 2; ++it)
            async_copy_b128(&As[s][(arow + it * 64) * LDK + ac8],
                            oge + (size_t)(m0 + arow + it * 64) * kH + kb + ac8);
#pragma unroll
        for (int q = 0; q < 4; ++q)
            b_st[q] = *(const v4f*)(wpe + (size_t)(kb + br + q) * kD + n0 + bn4);
    };
    auto stage_b = [&](int s) {
#pragma unroll
        for (int j = 0; j < 4; ++j) {   // k-quad: two packed dwords, 8B store
            uint2 r;
            r.x = pack2_bf16(b_st[0][j], b_st[1][j]);
            r.y = pack2_bf16(b_st[2][j], b_st[3][j]);
            *(uint2*)&Bs[s][(bn4 + j) * LDK + br] = r;
        }
    };

    v8f acc[4][2];
#pragma unroll
    for (int i = 0; i < 4; ++i)
#pragma unroll
        for (int j = 0; j < 2; ++j) acc[i][j] = (v8f)0.0f;

    fetch(0, 0);
    stage_b(0);
    wait_async();
    int s = 0;
    for (int kb = 0; kb < kH; kb += BK, s ^= 1) {
        __syncthreads();
        const bool more = (kb + BK) < kH;
        if (more) fetch(kb + BK, s ^ 1);   // async A into s^1 + B global loads

        v16bf afrag[4], bfrag[2];
#pragma unroll
        for (int i = 0; i < 4; ++i)
            afrag[i] = load_frag(&As[s][(wm * 64 + i * 16 + l16) * LDK], lhalf);
#pragma unroll
        for (int j = 0; j < 2; ++j)
            bfrag[j] = load_frag(&Bs[s][(wn * 32 + j * 16 + l16) * LDK], lhalf);
#pragma unroll
        for (int i = 0; i < 4; ++i)
#pragma unroll
            for (int j = 0; j < 2; ++j)
                acc[i][j] = wmma_bf16(afrag[i], bfrag[j], acc[i][j]);

        if (more) stage_b(s ^ 1);
        wait_async();   // A-tile for s^1 resident before publishing barrier
    }

    // ---- Epilogue: add bias, store f32 ----
#pragma unroll
    for (int j = 0; j < 2; ++j) {
        int col = n0 + wn * 32 + j * 16 + l16;
        float b = b_proj[(size_t)e * kD + col];
#pragma unroll
        for (int i = 0; i < 4; ++i) {
#pragma unroll
            for (int r = 0; r < 8; ++r) {
                int row = m0 + wm * 64 + i * 16 + lhalf * 8 + r;
                oe[(size_t)row * kD + col] = acc[i][j][r] + b;
            }
        }
    }
}

// ---------------------------------------------------------------------------
extern "C" void kernel_launch(void* const* d_in, const int* in_sizes, int n_in,
                              void* d_out, int out_size, void* d_ws, size_t ws_size,
                              hipStream_t stream) {
    const float* x        = (const float*)d_in[0];
    const float* w_c_fc   = (const float*)d_in[1];
    const float* b_c_fc   = (const float*)d_in[2];
    const float* w_gate   = (const float*)d_in[3];
    const float* b_gate   = (const float*)d_in[4];
    const float* w_c_proj = (const float*)d_in[5];
    const float* b_c_proj = (const float*)d_in[6];
    float* out = (float*)d_out;
    __bf16* og = (__bf16*)d_ws;   // E*T*H bf16 = 128 MB intermediate

    dim3 blk(256, 1, 1);
    dim3 g1(kT / 128, kH / 64, kE);   // 16 x 64 x 8 = 8192 workgroups
    swiglu_dualgemm_k1<<<g1, blk, 0, stream>>>(x, w_c_fc, b_c_fc,
                                               w_gate, b_gate, og);

    dim3 g2(kT / 128, kD / 128, kE);  // 16 x 8 x 8 = 1024 workgroups
    proj_gemm_k2<<<g2, blk, 0, stream>>>(og, w_c_proj, b_c_proj, out);
}
